// Block_51977694216642
// MI455X (gfx1250) — matrix operand order repaired and checked by reading, not compiled, not run on hardware
//
#include <hip/hip_runtime.h>

#define B_ 4
#define T_ 1024
#define C_ 512
#define H_ 4
#define D_ 128
#define M_ (B_*T_)      // 4096 token rows
#define F_ (4*C_)       // 2048 FFN hidden

#define AS1 __attribute__((address_space(1)))
#define AS3 __attribute__((address_space(3)))

typedef int v4i_t __attribute__((vector_size(16)));   // matches builtin param type

// CDNA5 async global->LDS DMA (ASYNCcnt-tracked), with sync fallback.
#if defined(__has_builtin)
#  if __has_builtin(__builtin_amdgcn_global_load_async_to_lds_b128)
#    define USE_ASYNC_LDS 1
#  endif
#endif
#ifndef USE_ASYNC_LDS
#  define USE_ASYNC_LDS 0
#endif

#if USE_ASYNC_LDS
#  if defined(__has_builtin) && __has_builtin(__builtin_amdgcn_s_wait_asynccnt)
#    define WAIT_ASYNC(n) __builtin_amdgcn_s_wait_asynccnt(n)
#  else
#    define WAIT_ASYNC(n) asm volatile("s_wait_asynccnt " #n ::: "memory")
#  endif
#  define ASYNC_CPY128(g, l) \
     __builtin_amdgcn_global_load_async_to_lds_b128((AS1 v4i_t*)(void*)(g), (AS3 v4i_t*)(void*)(l), 0, 0)
#else
#  define WAIT_ASYNC(n) ((void)0)
#endif

typedef __attribute__((ext_vector_type(16))) __bf16 v16bf;
typedef __attribute__((ext_vector_type(8)))  float  v8f;

union FragBF { v16bf v; unsigned u[8]; };

__device__ __forceinline__ unsigned short f2bf(float f) {
  unsigned u = __float_as_uint(f);
  u += 0x7fffu + ((u >> 16) & 1u);          // round-to-nearest-even
  return (unsigned short)(u >> 16);
}

// ---------------------------------------------------------------------------
// f32 -> bf16 elementwise convert (weights)
// ---------------------------------------------------------------------------
__global__ void cvt_bf16_kernel(const float* __restrict__ in,
                                unsigned short* __restrict__ out, int n) {
  int i = blockIdx.x * blockDim.x + threadIdx.x;
  if (i < n) out[i] = f2bf(in[i]);
}

// ---------------------------------------------------------------------------
// LayerNorm(x) + time-shift mix -> xq/xk/xv/xg (bf16).  One block per row.
// ---------------------------------------------------------------------------
__global__ __launch_bounds__(256) void ln1_shift_kernel(
    const float* __restrict__ x,
    const float* __restrict__ g1, const float* __restrict__ b1,
    const float* __restrict__ tmq, const float* __restrict__ tmk,
    const float* __restrict__ tmv, const float* __restrict__ tmg,
    unsigned short* __restrict__ xq, unsigned short* __restrict__ xk,
    unsigned short* __restrict__ xv, unsigned short* __restrict__ xg)
{
  __shared__ float r1[256], r2[256];
  const int row = blockIdx.x;
  const int t   = row & (T_ - 1);
  const int tid = threadIdx.x;
  const size_t base = (size_t)row * C_;

  float a0 = x[base + tid], a1 = x[base + tid + 256];
  r1[tid] = a0 + a1; r2[tid] = a0*a0 + a1*a1;
  __syncthreads();
  for (int st = 128; st > 0; st >>= 1) {
    if (tid < st) { r1[tid] += r1[tid+st]; r2[tid] += r2[tid+st]; }
    __syncthreads();
  }
  const float mc  = r1[0] * (1.f/C_);
  const float vc  = r2[0] * (1.f/C_) - mc*mc;
  const float rsc = rsqrtf(vc + 1e-5f);
  __syncthreads();

  float p0 = 0.f, p1 = 0.f, mp = 0.f, rsp = 0.f;
  const bool hasPrev = (t != 0);              // block-uniform
  if (hasPrev) {
    p0 = x[base - C_ + tid]; p1 = x[base - C_ + tid + 256];
    r1[tid] = p0 + p1; r2[tid] = p0*p0 + p1*p1;
    __syncthreads();
    for (int st = 128; st > 0; st >>= 1) {
      if (tid < st) { r1[tid] += r1[tid+st]; r2[tid] += r2[tid+st]; }
      __syncthreads();
    }
    mp = r1[0] * (1.f/C_);
    const float vp = r2[0] * (1.f/C_) - mp*mp;
    rsp = rsqrtf(vp + 1e-5f);
  }
  #pragma unroll
  for (int s = 0; s < 2; ++s) {
    const int c = tid + s*256;
    const float xc = (s == 0) ? a0 : a1;
    const float hc = g1[c]*(xc - mc)*rsc + b1[c];
    float hp = 0.f;
    if (hasPrev) {
      const float xp = (s == 0) ? p0 : p1;
      hp = g1[c]*(xp - mp)*rsp + b1[c];
    }
    xq[base + c] = f2bf(hc*tmq[c] + hp*(1.f - tmq[c]));
    xk[base + c] = f2bf(hc*tmk[c] + hp*(1.f - tmk[c]));
    xv[base + c] = f2bf(hc*tmv[c] + hp*(1.f - tmv[c]));
    xg[base + c] = f2bf(hc*tmg[c] + hp*(1.f - tmg[c]));
  }
}

// ---------------------------------------------------------------------------
// LayerNorm -> bf16 (no shift).  One block per row.
// ---------------------------------------------------------------------------
__global__ __launch_bounds__(256) void ln_bf16_kernel(
    const float* __restrict__ x, const float* __restrict__ g1,
    const float* __restrict__ b1, unsigned short* __restrict__ out)
{
  __shared__ float r1[256], r2[256];
  const int tid = threadIdx.x;
  const size_t base = (size_t)blockIdx.x * C_;
  float a0 = x[base + tid], a1 = x[base + tid + 256];
  r1[tid] = a0 + a1; r2[tid] = a0*a0 + a1*a1;
  __syncthreads();
  for (int st = 128; st > 0; st >>= 1) {
    if (tid < st) { r1[tid] += r1[tid+st]; r2[tid] += r2[tid+st]; }
    __syncthreads();
  }
  const float m  = r1[0] * (1.f/C_);
  const float v  = r2[0] * (1.f/C_) - m*m;
  const float rs = rsqrtf(v + 1e-5f);
  out[base + tid]       = f2bf(g1[tid]      *(a0 - m)*rs + b1[tid]);
  out[base + tid + 256] = f2bf(g1[tid + 256]*(a1 - m)*rs + b1[tid + 256]);
}

// ---------------------------------------------------------------------------
// q <- exp(rms_scale * q / (||q||/sqrt(C) + 1e-8)) per row, in place.
// ---------------------------------------------------------------------------
__global__ __launch_bounds__(256) void rmsexp_kernel(float* __restrict__ q,
                                                     const float* __restrict__ scale)
{
  __shared__ float r1[256];
  const int tid = threadIdx.x;
  const size_t base = (size_t)blockIdx.x * C_;
  float a0 = q[base + tid], a1 = q[base + tid + 256];
  r1[tid] = a0*a0 + a1*a1;
  __syncthreads();
  for (int st = 128; st > 0; st >>= 1) {
    if (tid < st) r1[tid] += r1[tid+st];
    __syncthreads();
  }
  const float norm = sqrtf(r1[0]) * 0.04419417382415922f;   // 1/sqrt(512)
  const float inv  = 1.f / (norm + 1e-8f);
  q[base + tid]       = expf(scale[tid]       * a0 * inv);
  q[base + tid + 256] = expf(scale[tid + 256] * a1 * inv);
}

__global__ void sigmoid_kernel(float* __restrict__ g, int n) {
  int i = blockIdx.x * blockDim.x + threadIdx.x;
  if (i < n) g[i] = 1.f / (1.f + expf(-g[i]));
}

__global__ void gelu_bf16_kernel(const float* __restrict__ in,
                                 unsigned short* __restrict__ out, int n) {
  int i = blockIdx.x * blockDim.x + threadIdx.x;
  if (i < n) {
    float v = in[i];
    out[i] = f2bf(0.5f * v * (1.f + erff(v * 0.7071067811865475f)));
  }
}

// ---------------------------------------------------------------------------
// Decayed linear-attention scan.  One block per (b,h); A[128x128] f32 stays
// resident in LDS (64KB -- needs CDNA5's 320KB/WGP LDS).  Emits g*(S/Z) as
// bf16, ready for the Wo WMMA GEMM.
// ---------------------------------------------------------------------------
__global__ __launch_bounds__(256) void scan_kernel(
    const float* __restrict__ q, const float* __restrict__ k,
    const float* __restrict__ v, const float* __restrict__ g,
    const float* __restrict__ tdec, unsigned short* __restrict__ attg)
{
  extern __shared__ float sm[];
  float* A  = sm;                 // D*D
  float* qs = A  + D_*D_;
  float* ks = qs + D_;
  float* vs = ks + D_;
  float* zz = vs + D_;
  float* lm = zz + D_;
  float* Sp = lm + D_;            // partial S from upper i-half
  float* Zt = Sp + D_;            // 1 float

  const int bh  = blockIdx.x;
  const int b   = bh >> 2;
  const int h   = bh & (H_ - 1);
  const int tid = threadIdx.x;
  const int j   = tid & (D_ - 1);
  const int ih  = tid >> 7;       // i-half: 0 or 1

  for (int i = tid; i < D_*D_; i += 256) A[i] = 0.f;
  if (tid < D_) {
    zz[tid] = 0.f;
    lm[tid] = expf(-expf(tdec[h*D_ + tid]));
  }
  __syncthreads();

  size_t base = ((size_t)b * T_) * C_ + (size_t)h * D_;
  for (int t = 0; t < T_; ++t, base += C_) {
    if (tid < D_) {                       // stage q,k,v and update z
      const float kv = k[base + tid];
      qs[tid] = q[base + tid];
      ks[tid] = kv;
      vs[tid] = v[base + tid];
      zz[tid] = lm[tid]*zz[tid] + kv;
    }
    __syncthreads();
    const float vj = vs[j];
    float part = 0.f;
    const int i0 = ih * 64;
    #pragma unroll 4
    for (int ii = 0; ii < 64; ++ii) {     // A[i][j]=lam_i*A+k_i*v_j ; S_j += q_i*A
      const int i = i0 + ii;
      const float a = fmaf(ks[i], vj, lm[i] * A[i*D_ + j]);
      A[i*D_ + j] = a;
      part = fmaf(qs[i], a, part);
    }
    if (ih) Sp[j] = part;
    if (tid < 32) {                       // Zt = q . z (wave32 shuffle reduce)
      float zp = 0.f;
      #pragma unroll
      for (int r = 0; r < 4; ++r) zp = fmaf(qs[tid + r*32], zz[tid + r*32], zp);
      #pragma unroll
      for (int msk = 16; msk > 0; msk >>= 1) zp += __shfl_xor(zp, msk, 32);
      if (tid == 0) Zt[0] = zp;
    }
    __syncthreads();
    if (tid < D_) {
      const float S = part + Sp[tid];
      attg[base + tid] = f2bf(g[base + tid] * (S / Zt[0]));
    }
  }
}

// ---------------------------------------------------------------------------
// bf16 WMMA GEMM (NT):  C[M,N] = A[M,K] * B[N,K]^T  (+ optional f32 residual).
// 256 threads = 8 wave32; block tile 128x64; each wave: 2x2 tiles of
// v_wmma_f32_16x16x32_bf16.  K slices double-buffered through LDS via
// GLOBAL_LOAD_ASYNC_TO_LDS_B128 (ASYNCcnt) when available.  LDS rows padded
// to 20 dwords: 16B-aligned for B128 and conflict-free fragment gathers.
// ---------------------------------------------------------------------------
#define LSTR 20

__global__ __launch_bounds__(256) void gemm_bf16_nt(
    const unsigned short* __restrict__ Abf,
    const unsigned short* __restrict__ Bbf,
    const float* __restrict__ res, float* __restrict__ Cout,
    int Mdim, int Ndim, int Kdim)
{
  __shared__ unsigned la[2][128 * LSTR];
  __shared__ unsigned lb[2][64 * LSTR];

  const int tid  = threadIdx.x;
  const int lane = tid & 31;
  const int wave = tid >> 5;
  const int wm   = wave & 3;          // 4 m sub-tiles of 32
  const int wn   = wave >> 2;         // 2 n sub-tiles of 32
  const int half = lane >> 4;
  const int l16  = lane & 15;
  const int bM   = blockIdx.y * 128;
  const int bN   = blockIdx.x * 64;

  const int srow = tid >> 2;          // staging row 0..63
  const int skc  = tid & 3;           // 16B chunk within a 32-bf16 row slice

  v8f acc[2][2];
  #pragma unroll
  for (int mt = 0; mt < 2; ++mt)
    #pragma unroll
    for (int nt = 0; nt < 2; ++nt)
      #pragma unroll
      for (int e = 0; e < 8; ++e) acc[mt][nt][e] = 0.f;

  const int nK = Kdim >> 5;

  // Stage one 32-wide K slice into LDS buffer bu: 3 x 16B per thread.
  auto stage = [&](int kb, int bu) {
    const int k0 = kb << 5;
    const unsigned short* gA0 = Abf + (size_t)(bM + srow)      * Kdim + k0 + skc*8;
    const unsigned short* gA1 = Abf + (size_t)(bM + srow + 64) * Kdim + k0 + skc*8;
    const unsigned short* gB  = Bbf + (size_t)(bN + srow)      * Kdim + k0 + skc*8;
    unsigned* dA0 = &la[bu][(srow)      * LSTR + skc*4];
    unsigned* dA1 = &la[bu][(srow + 64) * LSTR + skc*4];
    unsigned* dB  = &lb[bu][(srow)      * LSTR + skc*4];
#if USE_ASYNC_LDS
    ASYNC_CPY128(gA0, dA0);
    ASYNC_CPY128(gA1, dA1);
    ASYNC_CPY128(gB,  dB);
#else
    const uint4 a0 = *(const uint4*)gA0;      // issue all three loads, then store
    const uint4 a1 = *(const uint4*)gA1;
    const uint4 b0 = *(const uint4*)gB;
    *(uint4*)dA0 = a0;
    *(uint4*)dA1 = a1;
    *(uint4*)dB  = b0;
#endif
  };

  stage(0, 0);

  for (int kb = 0; kb < nK; ++kb) {
    const int bu = kb & 1;
    if (kb + 1 < nK) stage(kb + 1, bu ^ 1);        // overlap DMA with compute
    if (kb + 2 < nK) {                             // L2 prefetch 2 slices ahead
      const int pk = (kb + 2) << 5;
      __builtin_prefetch(Abf + (size_t)(bM + (tid & 127))*Kdim + pk, 0, 1);
      if (tid < 64)
        __builtin_prefetch(Bbf + (size_t)(bN + tid)*Kdim + pk, 0, 1);
    }
#if USE_ASYNC_LDS
    if (kb + 1 < nK) { WAIT_ASYNC(3); }            // retire oldest DMA group
    else             { WAIT_ASYNC(0); }
#endif
    __syncthreads();

    // ---- fragments (ISA 7.12.2 16-bit layouts)
    FragBF af[2], bfv[2];
    #pragma unroll
    for (int mt = 0; mt < 2; ++mt) {
      const int row = wm*32 + mt*16 + l16;        // A: M = lane%16
      #pragma unroll
      for (int p = 0; p < 8; ++p) {               // pair p -> K = base+half*8+2p'
        const int kp = ((p & 4) ? 16 : 0) + half*8 + (p & 3)*2;
        af[mt].u[p] = la[bu][row*LSTR + (kp >> 1)];
      }
    }
    #pragma unroll
    for (int nt = 0; nt < 2; ++nt) {
      const int rowb = wn*32 + nt*16 + l16;       // B: N = lane%16, K same order as A
      #pragma unroll
      for (int p = 0; p < 8; ++p)
        bfv[nt].u[p] = lb[bu][rowb*LSTR + ((p & 4) ? 8 : 0) + half*4 + (p & 3)];
    }
    #pragma unroll
    for (int mt = 0; mt < 2; ++mt)
      #pragma unroll
      for (int nt = 0; nt < 2; ++nt)
        acc[mt][nt] = __builtin_amdgcn_wmma_f32_16x16x32_bf16(
            false, af[mt].v, false, bfv[nt].v, (short)0, acc[mt][nt], false, false);
    __syncthreads();                               // buffer reuse fence
  }

  // ---- store (C layout: m = 8*half + e, n = lane%16), fused residual add
  #pragma unroll
  for (int mt = 0; mt < 2; ++mt)
    #pragma unroll
    for (int nt = 0; nt < 2; ++nt) {
      const int n = bN + wn*32 + nt*16 + l16;
      #pragma unroll
      for (int e = 0; e < 8; ++e) {
        const int m = bM + wm*32 + mt*16 + half*8 + e;
        const size_t idx = (size_t)m * Ndim + n;
        float vo = acc[mt][nt][e];
        if (res) vo += res[idx];
        Cout[idx] = vo;
      }
    }
}

// ---------------------------------------------------------------------------
extern "C" void kernel_launch(void* const* d_in, const int* in_sizes, int n_in,
                              void* d_out, int out_size, void* d_ws, size_t ws_size,
                              hipStream_t stream)
{
  (void)in_sizes; (void)n_in; (void)out_size; (void)ws_size;
  const float* x     = (const float*)d_in[0];
  const float* ln1_g = (const float*)d_in[1];
  const float* ln1_b = (const float*)d_in[2];
  const float* ln2_g = (const float*)d_in[3];
  const float* ln2_b = (const float*)d_in[4];
  const float* tdec  = (const float*)d_in[5];
  const float* tm_q  = (const float*)d_in[6];
  const float* tm_k  = (const float*)d_in[7];
  const float* tm_v  = (const float*)d_in[8];
  const float* tm_g  = (const float*)d_in[9];
  const float* Wq    = (const float*)d_in[10];
  const float* Wk    = (const float*)d_in[11];
  const float* Wv    = (const float*)d_in[12];
  const float* Wg    = (const float*)d_in[13];
  const float* Wo    = (const float*)d_in[14];
  const float* rms_s = (const float*)d_in[15];
  const float* W1    = (const float*)d_in[16];
  const float* W2    = (const float*)d_in[17];

  char* ws = (char*)d_ws;
  size_t o = 0;
  auto take = [&](size_t bytes) { size_t cur = o; o += (bytes + 255) & ~(size_t)255; return cur; };

  unsigned short* wq_b = (unsigned short*)(ws + take((size_t)C_*C_*2));
  unsigned short* wk_b = (unsigned short*)(ws + take((size_t)C_*C_*2));
  unsigned short* wv_b = (unsigned short*)(ws + take((size_t)C_*C_*2));
  unsigned short* wg_b = (unsigned short*)(ws + take((size_t)C_*C_*2));
  unsigned short* wo_b = (unsigned short*)(ws + take((size_t)C_*C_*2));
  unsigned short* w1_b = (unsigned short*)(ws + take((size_t)F_*C_*2));
  unsigned short* w2_b = (unsigned short*)(ws + take((size_t)C_*F_*2));
  // xq..xg contiguous (each a multiple of 256B) -> later reused for ffn act
  unsigned short* xq_b = (unsigned short*)(ws + take((size_t)M_*C_*2));
  unsigned short* xk_b = (unsigned short*)(ws + take((size_t)M_*C_*2));
  unsigned short* xv_b = (unsigned short*)(ws + take((size_t)M_*C_*2));
  unsigned short* xg_b = (unsigned short*)(ws + take((size_t)M_*C_*2));
  // qf..gf contiguous (32MB) -> later reused for ffn1 f32
  float* qf = (float*)(ws + take((size_t)M_*C_*4));
  float* kf = (float*)(ws + take((size_t)M_*C_*4));
  float* vf = (float*)(ws + take((size_t)M_*C_*4));
  float* gf = (float*)(ws + take((size_t)M_*C_*4));
  unsigned short* attg = (unsigned short*)(ws + take((size_t)M_*C_*2));
  float* x1            = (float*)(ws + take((size_t)M_*C_*4));
  unsigned short* h2b  = (unsigned short*)(ws + take((size_t)M_*C_*2));
  float* ffn1           = qf;                       // alias: 4x8MB span, free after scan
  unsigned short* ffact = xq_b;                     // alias: 4x4MB span, free after GEMMs

  const dim3 blk(256);

  // --- weights -> bf16
  cvt_bf16_kernel<<<(C_*C_+255)/256, blk, 0, stream>>>(Wq, wq_b, C_*C_);
  cvt_bf16_kernel<<<(C_*C_+255)/256, blk, 0, stream>>>(Wk, wk_b, C_*C_);
  cvt_bf16_kernel<<<(C_*C_+255)/256, blk, 0, stream>>>(Wv, wv_b, C_*C_);
  cvt_bf16_kernel<<<(C_*C_+255)/256, blk, 0, stream>>>(Wg, wg_b, C_*C_);
  cvt_bf16_kernel<<<(C_*C_+255)/256, blk, 0, stream>>>(Wo, wo_b, C_*C_);
  cvt_bf16_kernel<<<(F_*C_+255)/256, blk, 0, stream>>>(W1, w1_b, F_*C_);
  cvt_bf16_kernel<<<(C_*F_+255)/256, blk, 0, stream>>>(W2, w2_b, C_*F_);

  // --- LN1 + time-shift mix -> bf16 GEMM inputs
  ln1_shift_kernel<<<M_, blk, 0, stream>>>(x, ln1_g, ln1_b, tm_q, tm_k, tm_v, tm_g,
                                           xq_b, xk_b, xv_b, xg_b);

  // --- q,k,v,g projections (WMMA bf16, f32 out)
  const dim3 g512(C_/64, M_/128);
  gemm_bf16_nt<<<g512, blk, 0, stream>>>(xq_b, wq_b, nullptr, qf, M_, C_, C_);
  gemm_bf16_nt<<<g512, blk, 0, stream>>>(xk_b, wk_b, nullptr, kf, M_, C_, C_);
  gemm_bf16_nt<<<g512, blk, 0, stream>>>(xv_b, wv_b, nullptr, vf, M_, C_, C_);
  gemm_bf16_nt<<<g512, blk, 0, stream>>>(xg_b, wg_b, nullptr, gf, M_, C_, C_);

  sigmoid_kernel<<<(M_*C_+255)/256, blk, 0, stream>>>(gf, M_*C_);
  rmsexp_kernel<<<M_, blk, 0, stream>>>(qf, rms_s);
  rmsexp_kernel<<<M_, blk, 0, stream>>>(kf, rms_s);

  // --- sequential decayed scan, A resident in LDS; writes g*(S/Z) as bf16
  const size_t scanLds = (size_t)(D_*D_ + 6*D_ + 1) * sizeof(float);   // ~68.6KB
  scan_kernel<<<B_*H_, blk, scanLds, stream>>>(qf, kf, vf, gf, tdec, attg);

  // --- output projection with fused residual: x1 = x + attg @ Wo^T
  gemm_bf16_nt<<<g512, blk, 0, stream>>>(attg, wo_b, x, x1, M_, C_, C_);

  // --- ChannelMix
  ln_bf16_kernel<<<M_, blk, 0, stream>>>(x1, ln2_g, ln2_b, h2b);
  const dim3 gffn(F_/64, M_/128);
  gemm_bf16_nt<<<gffn, blk, 0, stream>>>(h2b, w1_b, nullptr, ffn1, M_, F_, C_);
  gelu_bf16_kernel<<<(M_*F_+255)/256, blk, 0, stream>>>(ffn1, ffact, M_*F_);
  // --- out = x1 + gelu @ W2^T
  gemm_bf16_nt<<<g512, blk, 0, stream>>>(ffact, w2_b, x1, (float*)d_out, M_, C_, F_);
}
